// LSTMModel_4183298146963
// MI455X (gfx1250) — compile-verified, matrix operand order checked by reading, not run on hardware
//
#include <hip/hip_runtime.h>

typedef __attribute__((ext_vector_type(16))) _Float16 v16h;
typedef __attribute__((ext_vector_type(8)))  float    v8f;

#define HID   48
#define GATES 192
#define TT    512
#define BBAT  2048
#define BLK_B 16   // batch columns per block (one N tile)

union HVec {
  v16h h;
  unsigned int u[8];
  uint4 q[2];
};

__device__ __forceinline__ unsigned int pack2(float a, float b) {
  union { _Float16 hh[2]; unsigned int u; } p;
  p.hh[0] = (_Float16)a;
  p.hh[1] = (_Float16)b;
  return p.u;
}

// CDNA5 has native V_TANH_F32 on the TRANS pipe; use it if the builtin exists.
#if __has_builtin(__builtin_amdgcn_tanhf)
__device__ __forceinline__ float gate_tanh(float x) {
  return __builtin_amdgcn_tanhf(x);
}
__device__ __forceinline__ float gate_sigmoid(float x) {
  // sig(x) = 0.5*tanh(0.5x) + 0.5 : mul + tanh + fma (1 TRANS op)
  return fmaf(0.5f, __builtin_amdgcn_tanhf(0.5f * x), 0.5f);
}
#else
__device__ __forceinline__ float gate_sigmoid(float x) {
  return __builtin_amdgcn_rcpf(1.0f + __builtin_amdgcn_exp2f(-1.44269504f * x));
}
__device__ __forceinline__ float gate_tanh(float x) {
  return 2.0f * __builtin_amdgcn_rcpf(1.0f + __builtin_amdgcn_exp2f(-2.88539008f * x)) - 1.0f;
}
#endif

__launch_bounds__(96, 1)
__global__ void lstm2_fused(const float* __restrict__ x,
                            const float* __restrict__ W_ih0,
                            const float* __restrict__ W_hh0,
                            const float* __restrict__ b_ih0,
                            const float* __restrict__ b_hh0,
                            const float* __restrict__ W_ih1,
                            const float* __restrict__ W_hh1,
                            const float* __restrict__ b_ih1,
                            const float* __restrict__ b_hh1,
                            const float* __restrict__ W_lin,
                            const float* __restrict__ b_lin,
                            float* __restrict__ out) {
  // Layer-0 augmented weights: [192 x 64] f16, cols 0..47 = W_hh0,
  // col 48 = W_ih0 (D=1), col 49 = b_ih0+b_hh0, rest zero.
  __shared__ __align__(16) _Float16 W0[GATES][64];
  // Layer-1 fused weights: [192 x 96] f16, cols 0..47 = W_ih1, 48..95 = W_hh1.
  __shared__ __align__(16) _Float16 W1[GATES][96];
  __shared__ __align__(16) float    B1[GATES];
  // Double-buffered hidden-state exchange, row = batch col, padded K pitch 64.
  __shared__ __align__(16) _Float16 HA[2][BLK_B][64];
  __shared__ __align__(16) _Float16 HB[2][BLK_B][64];

  const int tid   = threadIdx.x;       // 0..95, 3 waves
  const int wave  = tid >> 5;          // hidden block owned by this wave
  const int lane  = tid & 31;
  const int ln    = lane & 15;         // batch column within tile
  const int half  = (lane < 16) ? 0 : 8;
  const int bbase = blockIdx.x * BLK_B;

  // ---- stage weights into LDS (f32 -> f16) ----
  for (int i = tid; i < GATES * 64; i += 96) {
    int g = i >> 6, k = i & 63;
    float v = 0.0f;
    if (k < HID)       v = W_hh0[g * HID + k];
    else if (k == 48)  v = W_ih0[g];
    else if (k == 49)  v = b_ih0[g] + b_hh0[g];
    W0[g][k] = (_Float16)v;
  }
  for (int i = tid; i < GATES * 96; i += 96) {
    int g = i / 96, k = i - g * 96;
    float v = (k < HID) ? W_ih1[g * HID + k] : W_hh1[g * HID + (k - HID)];
    W1[g][k] = (_Float16)v;
  }
  for (int g = tid; g < GATES; g += 96) B1[g] = b_ih1[g] + b_hh1[g];
  {
    _Float16* ha = &HA[0][0][0];
    _Float16* hb = &HB[0][0][0];
    for (int i = tid; i < 2 * BLK_B * 64; i += 96) { ha[i] = (_Float16)0.0f; hb[i] = (_Float16)0.0f; }
  }
  __syncthreads();

  // ---- hoist A-matrix fragments (weights) into registers for the whole T loop ----
  // A 16-bit layout: lane<16 -> row 16*jm+ln, K base..base+7 (v0..3) and base+16..+23 (v4..7);
  // lane>=16 -> K base+8.. and base+24.. (base = 32*kf).
  HVec A0[4][2];   // layer 0: 4 tiles x 2 K-frags (K padded 50->64)
  HVec A1[4][3];   // layer 1: 4 tiles x 3 K-frags (K = 96)
  float bias1r[4][8];
#pragma unroll
  for (int jt = 0; jt < 4; ++jt) {
    const int row = 16 * (wave + 3 * jt) + ln;
#pragma unroll
    for (int kf = 0; kf < 2; ++kf) {
      const int kb = 32 * kf + half;
      A0[jt][kf].q[0] = *(const uint4*)&W0[row][kb];
      A0[jt][kf].q[1] = *(const uint4*)&W0[row][kb + 16];
    }
#pragma unroll
    for (int kf = 0; kf < 3; ++kf) {
      const int kb = 32 * kf + half;
      A1[jt][kf].q[0] = *(const uint4*)&W1[row][kb];
      A1[jt][kf].q[1] = *(const uint4*)&W1[row][kb + 16];
    }
    const int gb = 16 * (wave + 3 * jt) + half;
#pragma unroll
    for (int v = 0; v < 8; ++v) bias1r[jt][v] = B1[gb + v];
  }

  float cA[8], cB[8];
#pragma unroll
  for (int v = 0; v < 8; ++v) { cA[v] = 0.0f; cB[v] = 0.0f; }

  const float* xrow = x + (size_t)(bbase + ln) * TT;
  float4 xc = make_float4(0.f, 0.f, 0.f, 0.f);

#pragma unroll 2
  for (int t = 0; t < TT; ++t) {
    const int p = t & 1, q = p ^ 1;
    if ((t & 3) == 0) xc = *(const float4*)(xrow + t);   // uniform branch, 16B load / 4 steps
    const float xlo = (t & 1) ? xc.y : xc.x;
    const float xhi = (t & 1) ? xc.w : xc.z;
    const float xt  = (t & 2) ? xhi : xlo;

    // ================= layer 0 =================
    // B frag0 = h_a K0..31, frag1 = h_a K32..47 | {x_t, 1.0} at K48,49
    HVec bf0, bf1;
    bf0.q[0] = *(const uint4*)&HA[p][ln][half];
    bf0.q[1] = *(const uint4*)&HA[p][ln][16 + half];
    bf1.q[0] = *(const uint4*)&HA[p][ln][32 + half];
    bf1.u[4] = (lane < 16) ? pack2(xt, 1.0f) : 0u;
    bf1.u[5] = 0u; bf1.u[6] = 0u; bf1.u[7] = 0u;

    v8f acc[4];
#pragma unroll
    for (int jt = 0; jt < 4; ++jt) {
      v8f a;
#pragma unroll
      for (int v = 0; v < 8; ++v) a[v] = 0.0f;
      a = __builtin_amdgcn_wmma_f32_16x16x32_f16(false, A0[jt][0].h, false, bf0.h,
                                                 (short)0, a, false, false);
      a = __builtin_amdgcn_wmma_f32_16x16x32_f16(false, A0[jt][1].h, false, bf1.h,
                                                 (short)0, a, false, false);
      acc[jt] = a;
    }

    float hnew[8];
#pragma unroll
    for (int v = 0; v < 8; ++v) {
      float ig = gate_sigmoid(acc[0][v]);
      float fg = gate_sigmoid(acc[1][v]);
      float gg = gate_tanh(acc[2][v]);
      float og = gate_sigmoid(acc[3][v]);
      cA[v] = fg * cA[v] + ig * gg;
      hnew[v] = og * gate_tanh(cA[v]);
    }
    {
      uint4 w4;
      w4.x = pack2(hnew[0], hnew[1]);
      w4.y = pack2(hnew[2], hnew[3]);
      w4.z = pack2(hnew[4], hnew[5]);
      w4.w = pack2(hnew[6], hnew[7]);
      *(uint4*)&HA[q][ln][16 * wave + half] = w4;
    }
    __syncthreads();   // h_a(t) published by all 3 waves

    // ================= layer 1 =================
    // B vector = [h_a_new (K0..47) ; h_b_prev (K48..95)], 3 exact K-frags
    HVec f0, f1, f2;
    f0.q[0] = *(const uint4*)&HA[q][ln][half];
    f0.q[1] = *(const uint4*)&HA[q][ln][16 + half];
    f1.q[0] = *(const uint4*)&HA[q][ln][32 + half];
    f1.q[1] = *(const uint4*)&HB[p][ln][half];
    f2.q[0] = *(const uint4*)&HB[p][ln][16 + half];
    f2.q[1] = *(const uint4*)&HB[p][ln][32 + half];

#pragma unroll
    for (int jt = 0; jt < 4; ++jt) {
      v8f a;
#pragma unroll
      for (int v = 0; v < 8; ++v) a[v] = bias1r[jt][v];
      a = __builtin_amdgcn_wmma_f32_16x16x32_f16(false, A1[jt][0].h, false, f0.h,
                                                 (short)0, a, false, false);
      a = __builtin_amdgcn_wmma_f32_16x16x32_f16(false, A1[jt][1].h, false, f1.h,
                                                 (short)0, a, false, false);
      a = __builtin_amdgcn_wmma_f32_16x16x32_f16(false, A1[jt][2].h, false, f2.h,
                                                 (short)0, a, false, false);
      acc[jt] = a;
    }

#pragma unroll
    for (int v = 0; v < 8; ++v) {
      float ig = gate_sigmoid(acc[0][v]);
      float fg = gate_sigmoid(acc[1][v]);
      float gg = gate_tanh(acc[2][v]);
      float og = gate_sigmoid(acc[3][v]);
      cB[v] = fg * cB[v] + ig * gg;
      hnew[v] = og * gate_tanh(cB[v]);
    }
    {
      uint4 w4;
      w4.x = pack2(hnew[0], hnew[1]);
      w4.y = pack2(hnew[2], hnew[3]);
      w4.z = pack2(hnew[4], hnew[5]);
      w4.w = pack2(hnew[6], hnew[7]);
      *(uint4*)&HB[q][ln][16 * wave + half] = w4;
    }
    __syncthreads();   // h_b(t) published
  }

  // ---- final linear on h_b[T-1] (lives in HB[TT & 1]) ----
  if (wave == 0 && lane < 16) {
    const int b = bbase + lane;
    float accv = b_lin[0];
#pragma unroll 8
    for (int k = 0; k < HID; ++k)
      accv += (float)HB[TT & 1][lane][k] * W_lin[k];
    out[b] = accv;
  }
}

extern "C" void kernel_launch(void* const* d_in, const int* in_sizes, int n_in,
                              void* d_out, int out_size, void* d_ws, size_t ws_size,
                              hipStream_t stream) {
  (void)in_sizes; (void)n_in; (void)d_ws; (void)ws_size; (void)out_size;
  const float* x     = (const float*)d_in[0];
  const float* W_ih0 = (const float*)d_in[1];
  const float* W_hh0 = (const float*)d_in[2];
  const float* b_ih0 = (const float*)d_in[3];
  const float* b_hh0 = (const float*)d_in[4];
  const float* W_ih1 = (const float*)d_in[5];
  const float* W_hh1 = (const float*)d_in[6];
  const float* b_ih1 = (const float*)d_in[7];
  const float* b_hh1 = (const float*)d_in[8];
  const float* W_lin = (const float*)d_in[9];
  const float* b_lin = (const float*)d_in[10];
  float* out = (float*)d_out;

  lstm2_fused<<<dim3(BBAT / BLK_B), dim3(96), 0, stream>>>(
      x, W_ih0, W_hh0, b_ih0, b_hh0,
      W_ih1, W_hh1, b_ih1, b_hh1,
      W_lin, b_lin, out);
}
